// DynamicUnroll_31628139168613
// MI455X (gfx1250) — matrix operand order repaired and checked by reading, not compiled, not run on hardware
//
#include <hip/hip_runtime.h>
#include <math.h>

// Problem constants (from reference): B=64, T=1024, D=512, U=512
#define BB 64
#define TT 1024
#define DD 512
#define UU 512
#define NTOT 2048      // 4*U
#define KTOT 1024      // D + U
#define NKC  32        // K-chunks of 32 (KTOT/32)
#define NKC_HALF 16    // per-wave K-chunks after K-split
#define N_UTILE 32     // UU/16
#define NBLOCKS 32     // persistent grid: one block per utile

typedef __attribute__((ext_vector_type(16))) __bf16 v16bf;
typedef __attribute__((ext_vector_type(8)))  __bf16 v8bf;
typedef __attribute__((ext_vector_type(4)))  __bf16 v4bf;
typedef __attribute__((ext_vector_type(8)))  float  v8f;

__device__ __forceinline__ v16bf concat8(v8bf lo, v8bf hi) {
    return __builtin_shufflevector(lo, hi, 0,1,2,3,4,5,6,7,8,9,10,11,12,13,14,15);
}

__device__ __forceinline__ float fast_sigmoid(float x) {
    float e = __expf(-x);
    return __builtin_amdgcn_rcpf(1.0f + e);
}
__device__ __forceinline__ float fast_tanh(float x) {
    // tanh(x) = 1 - 2/(exp(2x)+1); saturates correctly to +/-1
    float e = __expf(2.0f * x);
    return 1.0f - 2.0f * __builtin_amdgcn_rcpf(e + 1.0f);
}

// ---------------------------------------------------------------------------
// Pack W = [Wx ; Wh] (fp32, row-major KTOT x NTOT) into per-wave WMMA
// B-fragment layout (bf16). Fragment (ntile, kc) is 512 contiguous bf16:
//   flat = ((ntile*NKC + kc) * 512) + lane*16 + slot
// semantic: slot e at lane L  <->  B[kc*32 + (e/8)*16 + (L/16)*8 + (e%8)][ntile*16 + (L%16)]
// ---------------------------------------------------------------------------
__global__ void pack_w_kernel(const float* __restrict__ Wx,
                              const float* __restrict__ Wh,
                              __bf16* __restrict__ Wpack) {
    int tid = blockIdx.x * 256 + threadIdx.x;      // 0 .. 2097151
    int e     = tid & 15;
    int L     = (tid >> 4) & 31;
    int kc    = (tid >> 9) & 31;
    int ntile = tid >> 14;
    int K = kc * 32 + ((e >> 3) << 4) + ((L >> 4) << 3) + (e & 7);
    int N = ntile * 16 + (L & 15);
    float v = (K < DD) ? Wx[(size_t)K * NTOT + N]
                       : Wh[(size_t)(K - DD) * NTOT + N];
    Wpack[tid] = (__bf16)v;
}

// x fp32 [B,T,D] -> bf16 same layout (4 elements per thread)
__global__ void cvt_x_kernel(const float* __restrict__ x,
                             __bf16* __restrict__ xb) {
    size_t i = ((size_t)blockIdx.x * 256 + threadIdx.x) * 4;
    float4 v = *(const float4*)(x + i);
    v4bf o = { (__bf16)v.x, (__bf16)v.y, (__bf16)v.z, (__bf16)v.w };
    *(v4bf*)(xb + i) = o;
}

// zero c_state (fp32 B*U), h0 (bf16 B*U), and the grid-barrier words.
// reset_mask[:,0] is all-true in the reference, so zero initial state is exact.
__global__ void init_state_kernel(float* __restrict__ c_state,
                                  unsigned* __restrict__ h0_u32,
                                  unsigned* __restrict__ bar) {
    int i = blockIdx.x * 256 + threadIdx.x;        // 0 .. 32767
    c_state[i] = 0.0f;
    if (i < (BB * UU) / 2) h0_u32[i] = 0u;         // 16384 dwords of bf16 h0
    if (i < 2) bar[i] = 0u;                        // {count, generation}
}

// device-scope grid barrier: one release/acquire round per step
__device__ __forceinline__ void grid_sync(unsigned* cnt, unsigned* gen) {
    __syncthreads();
    if (threadIdx.x == 0) {
        unsigned g = __hip_atomic_load(gen, __ATOMIC_RELAXED,
                                       __HIP_MEMORY_SCOPE_AGENT);
        unsigned arrived = __hip_atomic_fetch_add(cnt, 1u, __ATOMIC_ACQ_REL,
                                                  __HIP_MEMORY_SCOPE_AGENT);
        if (arrived == NBLOCKS - 1) {
            __hip_atomic_store(cnt, 0u, __ATOMIC_RELAXED,
                               __HIP_MEMORY_SCOPE_AGENT);
            __hip_atomic_fetch_add(gen, 1u, __ATOMIC_ACQ_REL,
                                   __HIP_MEMORY_SCOPE_AGENT);
        } else {
            while (__hip_atomic_load(gen, __ATOMIC_ACQUIRE,
                                     __HIP_MEMORY_SCOPE_AGENT) == g) {
                __builtin_amdgcn_s_sleep(2);
            }
        }
    }
    __syncthreads();
}

// ---------------------------------------------------------------------------
// Persistent LSTM scan. 32 blocks x 256 threads = 256 wave32s, one block per
// utile. Each block stages its 128 KB of WMMA B-fragments (4 gates x 32
// K-chunks) into LDS ONCE, then runs all 1024 time steps feeding WMMA from
// ds_load_b128, with one device-scope barrier per step (no per-step launches).
// Per tile (mtile=wave/2): khalf=0 wave does x@Wx (kc 0..15), khalf=1 wave
// does h@Wh (kc 16..31); partials combined via LDS. Reset masking is applied
// on the WRITE side (step t stores h/c pre-masked with reset_mask[:,t+1]),
// so the GEMM A-loads are branchless.
// ---------------------------------------------------------------------------
__global__ void __launch_bounds__(256)
lstm_scan_kernel(const __bf16* __restrict__ Xbf,          // [B,T,D] bf16
                 const unsigned char* __restrict__ rmask, // [B,T] bool
                 const __bf16* __restrict__ Wpack,        // packed B frags
                 const float* __restrict__ bias,          // [4U]
                 __bf16* __restrict__ h0,                 // [B,U] bf16 (even t in)
                 __bf16* __restrict__ h1,                 // [B,U] bf16 (odd t in)
                 float* __restrict__ c_state,             // [B,U] fp32
                 float* __restrict__ y_out,               // [B,T,U] fp32
                 float* __restrict__ hc_fin,              // [2*B*U] finals
                 unsigned* __restrict__ bar) {
    __shared__ __bf16 Blds[4 * NKC * 512];  // 128 KB: [gate][kc][512]
    __shared__ float  red[4 * 4 * 256];     // 16 KB reduction scratch

    const int lane  = threadIdx.x & 31;
    const int w     = threadIdx.x >> 5;        // 0..7
    const int mtile = w >> 1;                  // 0..3
    const int khalf = w & 1;                   // which K half
    const int utile = blockIdx.x;              // 0..31
    const int half  = lane >> 4;               // 0/1
    const int ln    = lane & 15;

    // ---- stage this block's B-fragments into LDS (one time) ----
#pragma unroll
    for (int g = 0; g < 4; ++g) {
        const uint4* src = (const uint4*)(Wpack +
            (((size_t)(g * N_UTILE + utile) * NKC) << 9));
        uint4* dst = (uint4*)(Blds + (g << 14));      // g * 32*512
        for (int i = threadIdx.x; i < 2048; i += 256) // 32 KB per gate
            dst[i] = src[i];
    }
    __syncthreads();

    // per-lane invariants
    const int ma = mtile * 16 + ln;                    // A-row for GEMM loads
    const __bf16* xrow0 = Xbf + (size_t)ma * TT * DD;  // + t*DD per step
    const __bf16* hrow0 = h0 + (size_t)ma * UU;
    const __bf16* hrow1 = h1 + (size_t)ma * UU;
    const __bf16* blane = Blds + ((size_t)khalf * NKC_HALF << 9) + lane * 16;
    float* rt = red + (mtile << 10) + lane * 8;        // per-gate: +g*256

    const int u = utile * 16 + ln;                     // C/D col (khalf=0 use)
    const float bi  = bias[u];
    const float bf_ = bias[UU + u];
    const float bg  = bias[2 * UU + u];
    const float bo  = bias[3 * UU + u];

    for (int t = 0; t < TT; ++t) {
        const __bf16* arow = (khalf == 0) ? (xrow0 + (size_t)t * DD)
                                          : ((t & 1) ? hrow1 : hrow0);
        v8f acc[4] = {};   // gates i, f, g, o

        // prologue: fragments for first K-chunk of this wave's half
        v8bf lo = *(const v8bf*)(arow + half * 8);
        v8bf hi = *(const v8bf*)(arow + half * 8 + 16);
        v16bf a = concat8(lo, hi);
        v16bf bcur[4];
#pragma unroll
        for (int g = 0; g < 4; ++g)
            bcur[g] = *(const v16bf*)(blane + (g << 14));

        // pipelined main loop: issue it+1 loads before it's WMMAs
#pragma unroll
        for (int it = 0; it < NKC_HALF - 1; ++it) {
            const __bf16* p = arow + (it + 1) * 32 + half * 8;
            v8bf nlo = *(const v8bf*)(p);
            v8bf nhi = *(const v8bf*)(p + 16);
            v16bf anext = concat8(nlo, nhi);
            v16bf bnext[4];
#pragma unroll
            for (int g = 0; g < 4; ++g)
                bnext[g] = *(const v16bf*)(blane + (g << 14) +
                                           ((it + 1) << 9));
#pragma unroll
            for (int g = 0; g < 4; ++g)
                acc[g] = __builtin_amdgcn_wmma_f32_16x16x32_bf16(
                             false, a, false, bcur[g], (short)0, acc[g],
                             false, false);
            a = anext;
#pragma unroll
            for (int g = 0; g < 4; ++g) bcur[g] = bnext[g];
        }
#pragma unroll
        for (int g = 0; g < 4; ++g)   // epilogue WMMAs (last K-chunk)
            acc[g] = __builtin_amdgcn_wmma_f32_16x16x32_bf16(
                         false, a, false, bcur[g], (short)0, acc[g],
                         false, false);

        // ---- combine the two K-halves through LDS ----
        if (khalf == 1) {
#pragma unroll
            for (int g = 0; g < 4; ++g) *(v8f*)(rt + g * 256) = acc[g];
        }
        __syncthreads();

        if (khalf == 0) {
#pragma unroll
            for (int g = 0; g < 4; ++g) {
                v8f other = *(const v8f*)(rt + g * 256);
#pragma unroll
                for (int j = 0; j < 8; ++j) acc[g][j] += other[j];
            }

            // ---- pointwise LSTM update, write-side reset masking ----
            __bf16* hout = (t & 1) ? h0 : h1;   // buffer read at step t+1
            const bool last = (t == TT - 1);
#pragma unroll
            for (int j = 0; j < 8; ++j) {
                const int m = mtile * 16 + half * 8 + j;
                const size_t cu = (size_t)m * UU + u;
                const float c_old = c_state[cu];

                const float zi = acc[0][j] + bi;
                const float zf = acc[1][j] + bf_;
                const float zg = acc[2][j] + bg;
                const float zo = acc[3][j] + bo;

                const float ig = fast_sigmoid(zi);
                const float fg = fast_sigmoid(zf);
                const float gg = fast_tanh(zg);
                const float og = fast_sigmoid(zo);

                const float c_new = fg * c_old + ig * gg;
                const float h_new = og * fast_tanh(c_new);

                // mask with NEXT step's reset so step t+1 loads branchlessly
                const bool rstn = !last && (rmask[(size_t)m * TT + t + 1] != 0);
                c_state[cu] = rstn ? 0.0f : c_new;
                hout[cu]    = (__bf16)(rstn ? 0.0f : h_new);
                y_out[((size_t)m * TT + t) * UU + u] = h_new;
                if (last) {                    // append h_f, c_f (unmasked)
                    hc_fin[cu] = h_new;
                    hc_fin[(size_t)BB * UU + cu] = c_new;
                }
            }
        }

        grid_sync(bar, bar + 1);
    }
}

// ---------------------------------------------------------------------------
extern "C" void kernel_launch(void* const* d_in, const int* in_sizes, int n_in,
                              void* d_out, int out_size, void* d_ws, size_t ws_size,
                              hipStream_t stream) {
    const float*         x     = (const float*)d_in[0];
    const unsigned char* rmask = (const unsigned char*)d_in[1];
    const float*         Wx    = (const float*)d_in[2];
    const float*         Wh    = (const float*)d_in[3];
    const float*         bias  = (const float*)d_in[4];

    // workspace layout (all 256B-aligned):
    //   Wpack : KTOT*NTOT bf16            = 4 MiB
    //   Xbf   : B*T*D bf16                = 64 MiB
    //   c     : B*U fp32                  = 128 KiB
    //   h0,h1 : B*U bf16 each             = 64 KiB each
    //   bar   : 2 x u32 (grid barrier)
    const size_t OFF_W  = 0;
    const size_t OFF_X  = OFF_W + (size_t)KTOT * NTOT * 2;
    const size_t OFF_C  = OFF_X + (size_t)BB * TT * DD * 2;
    const size_t OFF_H0 = OFF_C + (size_t)BB * UU * 4;
    const size_t OFF_H1 = OFF_H0 + (size_t)BB * UU * 2;
    const size_t OFF_B  = OFF_H1 + (size_t)BB * UU * 2;
    const size_t NEED   = OFF_B + 256;
    if (ws_size < NEED) return;  // insufficient scratch; cannot compute

    char* ws = (char*)d_ws;
    __bf16*   Wpack  = (__bf16*)(ws + OFF_W);
    __bf16*   Xbf    = (__bf16*)(ws + OFF_X);
    float*    cstate = (float*)(ws + OFF_C);
    __bf16*   h0     = (__bf16*)(ws + OFF_H0);
    __bf16*   h1     = (__bf16*)(ws + OFF_H1);
    unsigned* bar    = (unsigned*)(ws + OFF_B);

    float* out    = (float*)d_out;
    float* hc_fin = out + (size_t)BB * TT * UU;   // h_f then c_f

    // one-time (per launch) prep — fully parallel
    pack_w_kernel<<<(KTOT * NTOT) / 256, 256, 0, stream>>>(Wx, Wh, Wpack);
    cvt_x_kernel<<<((size_t)BB * TT * DD / 4) / 256, 256, 0, stream>>>(x, Xbf);
    init_state_kernel<<<(BB * UU) / 256, 256, 0, stream>>>(
        cstate, (unsigned*)h0, bar);

    // persistent scan: all 1024 steps in one kernel, LDS-resident weights,
    // one device-scope barrier per step.
    lstm_scan_kernel<<<NBLOCKS, 256, 0, stream>>>(
        Xbf, rmask, Wpack, bias, h0, h1, cstate, out, hc_fin, bar);
}
